// Int8LlamaDecoderLayer_16338055594000
// MI455X (gfx1250) — 16-bit and 8-bit matrix operand order repaired and checked by reading, compiled, not run on hardware
//
#include <hip/hip_runtime.h>
#include <hip/hip_fp16.h>

#define S_LEN 2048
#define HS 4096
#define NH 32
#define HD 128
#define IM 11008

typedef __attribute__((ext_vector_type(2)))  int      v2i;
typedef __attribute__((ext_vector_type(4)))  int      vi4;
typedef __attribute__((ext_vector_type(8)))  int      v8i;
typedef __attribute__((ext_vector_type(8)))  float    v8f;
typedef __attribute__((ext_vector_type(16))) _Float16 v16h;

#if __has_builtin(__builtin_amdgcn_global_load_async_to_lds_b128)
#define HAVE_ASYNC_LDS 1
typedef __attribute__((address_space(1))) vi4* gvi4p;
typedef __attribute__((address_space(3))) vi4* lvi4p;
static __device__ __forceinline__ void async_ld16(const void* g, void* l) {
  __builtin_amdgcn_global_load_async_to_lds_b128(
      (gvi4p)(unsigned long long)g,
      (lvi4p)(unsigned)(unsigned long long)l, 0, 0);
}
#else
#define HAVE_ASYNC_LDS 0
#endif

static __device__ __forceinline__ void wait_async0() {
#if __has_builtin(__builtin_amdgcn_s_wait_asynccnt)
  __builtin_amdgcn_s_wait_asynccnt(0);
#else
  asm volatile("s_wait_asynccnt 0" ::: "memory");
#endif
}

__global__ __launch_bounds__(256) void k_quant_w(const float* __restrict__ w,
                                                 signed char* __restrict__ o, int n4) {
  int gid = blockIdx.x * 256 + threadIdx.x;
  if (gid >= n4) return;
  int base = gid * 4, pack = 0;
#pragma unroll
  for (int j = 0; j < 4; ++j) {
    float q = rintf(w[base + j]);
    q = fminf(fmaxf(q, -128.f), 127.f);
    pack |= ((int)q & 255) << (8 * j);
  }
  *(int*)&o[base] = pack;
}

__global__ __launch_bounds__(256) void k_rmsnorm_q(const float* __restrict__ x,
                                                   const float* __restrict__ w,
                                                   signed char* __restrict__ o) {
  const int row = blockIdx.x;
  const float* xr = x + (size_t)row * HS;
  float ss = 0.f;
  for (int i = threadIdx.x; i < HS; i += 256) { float v = xr[i]; ss += v * v; }
#pragma unroll
  for (int m = 16; m >= 1; m >>= 1) ss += __shfl_xor(ss, m, 32);
  __shared__ float red[8];
  __shared__ float rstd;
  if ((threadIdx.x & 31) == 0) red[threadIdx.x >> 5] = ss;
  __syncthreads();
  if (threadIdx.x == 0) {
    float t = 0.f;
    for (int i = 0; i < 8; ++i) t += red[i];
    rstd = rsqrtf(t / (float)HS + 1e-6f);
  }
  __syncthreads();
  const float rs = rstd;
  for (int i = threadIdx.x; i < HS; i += 256) {
    float y = rintf(w[i] * xr[i] * rs);
    y = fminf(fmaxf(y, -128.f), 127.f);
    o[(size_t)row * HS + i] = (signed char)y;
  }
}

enum { EPI_F16 = 0, EPI_F32 = 1, EPI_RES = 2, EPI_SILU = 3 };

template <int EPI>
__global__ __launch_bounds__(256) void k_gemm_i8(const signed char* __restrict__ X,
                                                 const signed char* __restrict__ W,
                                                 const float* __restrict__ bias,
                                                 const float* __restrict__ alpha_p,
                                                 const float* __restrict__ res,
                                                 void* __restrict__ out,
                                                 int M, int N, int K) {
  constexpr int BK = 128;
  __shared__ __align__(16) signed char As[2][128 * BK];
  __shared__ __align__(16) signed char Bs[2][128 * BK];
  const int tid  = threadIdx.x;
  const int lane = tid & 31;
  const int wave = tid >> 5;
  const int l16  = lane & 15;
  const bool hi  = lane >= 16;
  const int bM = blockIdx.y * 128;
  const int bN = blockIdx.x * 128;
  const int wM = (wave >> 2) * 64;
  const int wN = (wave & 3) * 32;

  v8i acc[4][2];
#pragma unroll
  for (int mf = 0; mf < 4; ++mf)
#pragma unroll
    for (int nf = 0; nf < 2; ++nf)
#pragma unroll
      for (int r = 0; r < 8; ++r) acc[mf][nf][r] = 0;

  auto stage = [&](int buf, int k0) {
#pragma unroll
    for (int j = 0; j < 4; ++j) {
      const int i = tid + 256 * j;
      const int row = i >> 3;
      const int c16 = (i & 7) * 16;
      const signed char* gx = &X[(size_t)(bM + row) * K + k0 + c16];
      const signed char* gw = &W[(size_t)(bN + row) * K + k0 + c16];
#if HAVE_ASYNC_LDS
      async_ld16(gx, &As[buf][row * BK + c16]);
      async_ld16(gw, &Bs[buf][row * BK + c16]);
#else
      *(vi4*)&As[buf][row * BK + c16] = *(const vi4*)gx;
      *(vi4*)&Bs[buf][row * BK + c16] = *(const vi4*)gw;
#endif
    }
    if (k0 + BK < K) {
      __builtin_prefetch(&X[(size_t)(bM + (tid >> 1)) * K + k0 + BK], 0, 1);
      __builtin_prefetch(&W[(size_t)(bN + (tid >> 1)) * K + k0 + BK], 0, 1);
    }
  };

  const int T = K / BK;
  stage(0, 0);
  wait_async0();
  __syncthreads();
  for (int t = 0; t < T; ++t) {
    if (t + 1 < T) stage((t + 1) & 1, (t + 1) * BK);
    const signed char* Ab = As[t & 1];
    const signed char* Bb = Bs[t & 1];
#pragma unroll
    for (int kk = 0; kk < BK; kk += 64) {
      v8i a[4];
#pragma unroll
      for (int mf = 0; mf < 4; ++mf) {
        const signed char* p = &Ab[(wM + mf * 16 + l16) * BK + kk + (hi ? 16 : 0)];
        union { v8i v; v2i d[4]; } ua;
        ua.d[0] = *(const v2i*)(p);
        ua.d[1] = *(const v2i*)(p + 8);
        ua.d[2] = *(const v2i*)(p + 32);
        ua.d[3] = *(const v2i*)(p + 40);
        a[mf] = ua.v;
      }
      v8i b[2];
#pragma unroll
      for (int nf = 0; nf < 2; ++nf) {
        const signed char* p = &Bb[(wN + nf * 16 + l16) * BK + kk + (hi ? 16 : 0)];
        union { v8i v; vi4 q[2]; } ub;
        ub.q[0] = *(const vi4*)(p);
        ub.q[1] = *(const vi4*)(p + 32);
        b[nf] = ub.v;
      }
#pragma unroll
      for (int mf = 0; mf < 4; ++mf)
#pragma unroll
        for (int nf = 0; nf < 2; ++nf)
          acc[mf][nf] = __builtin_amdgcn_wmma_i32_16x16x64_iu8(
              true, a[mf], true, b[nf], acc[mf][nf], false, false);
    }
    wait_async0();
    __syncthreads();
  }

  const float alpha = *alpha_p;
#pragma unroll
  for (int mf = 0; mf < 4; ++mf)
#pragma unroll
    for (int nf = 0; nf < 2; ++nf) {
      const int n = bN + wN + nf * 16 + l16;
      const float bn = bias[n];
#pragma unroll
      for (int r = 0; r < 8; ++r) {
        const int m = bM + wM + mf * 16 + r + (hi ? 8 : 0);
        const size_t idx = (size_t)m * N + n;
        const float v = alpha * (float)acc[mf][nf][r] + bn;
        if (EPI == EPI_F16) {
          ((_Float16*)out)[idx] = (_Float16)v;
        } else if (EPI == EPI_F32) {
          ((float*)out)[idx] = v;
        } else if (EPI == EPI_RES) {
          ((float*)out)[idx] = res[idx] + v;
        } else {
          const float g  = (float)(_Float16)v;
          const float sg = g / (1.f + __expf(-g));
          ((_Float16*)out)[idx] = (_Float16)sg;
        }
      }
    }
}

__global__ __launch_bounds__(256) void k_rope(_Float16* __restrict__ q,
                                              _Float16* __restrict__ k) {
  const int gid = blockIdx.x * 256 + threadIdx.x;
  const int i = gid & 63;
  const int h = (gid >> 6) & (NH - 1);
  const int s = gid >> 11;
  const float ang = (float)s * __expf(-(float)i * 0.14391156831f);
  const _Float16 c  = (_Float16)cosf(ang);
  const _Float16 sn = (_Float16)sinf(ang);
  const size_t b = ((size_t)s * NH + h) * HD + i;
  _Float16 q1 = q[b], q2 = q[b + 64];
  q[b]      = q1 * c - q2 * sn;
  q[b + 64] = q2 * c + q1 * sn;
  _Float16 k1 = k[b], k2 = k[b + 64];
  k[b]      = k1 * c - k2 * sn;
  k[b + 64] = k2 * c + k1 * sn;
}

__global__ __launch_bounds__(256) void k_vtrans(const _Float16* __restrict__ v,
                                                _Float16* __restrict__ vt) {
  const int gid = blockIdx.x * 256 + threadIdx.x;
  const int d = gid & (HD - 1);
  const int h = (gid >> 7) & (NH - 1);
  const int s = gid >> 12;
  vt[((size_t)h * HD + d) * S_LEN + s] = v[gid];
}

__global__ __launch_bounds__(32) void k_attn(const _Float16* __restrict__ Q,
                                             const _Float16* __restrict__ Km,
                                             const _Float16* __restrict__ VT,
                                             _Float16* __restrict__ O) {
  __shared__ __align__(16) _Float16 Ps[16 * 32];
  const int lane = threadIdx.x;
  const int l16  = lane & 15;
  const bool hi  = lane >= 16;
  const int h = blockIdx.y;
  const int qBase = blockIdx.x * 16;
  const float scale = 0.08838834764831845f;

  v16h aQ[4];
#pragma unroll
  for (int dc = 0; dc < 4; ++dc) {
    const _Float16* p = Q + (size_t)(qBase + l16) * HS + h * HD + dc * 32 + (hi ? 8 : 0);
    union { v16h v; vi4 i[2]; } u;
    u.i[0] = *(const vi4*)p;
    u.i[1] = *(const vi4*)(p + 16);
    aQ[dc] = u.v;
  }

  float mrow[8], lrow[8];
  v8f accO[8];
#pragma unroll
  for (int r = 0; r < 8; ++r) { mrow[r] = -3.0e38f; lrow[r] = 0.f; }
#pragma unroll
  for (int dc = 0; dc < 8; ++dc)
#pragma unroll
    for (int r = 0; r < 8; ++r) accO[dc][r] = 0.f;

  const int ktEnd = (qBase + 16 + 31) >> 5;
  for (int kt = 0; kt < ktEnd; ++kt) {
    const int keyBase = kt * 32;
    v8f c0, c1;
#pragma unroll
    for (int r = 0; r < 8; ++r) { c0[r] = 0.f; c1[r] = 0.f; }
#pragma unroll
    for (int nf = 0; nf < 2; ++nf) {
      const _Float16* kp =
          Km + (size_t)(keyBase + nf * 16 + l16) * HS + h * HD + (hi ? 8 : 0);
#pragma unroll
      for (int dc = 0; dc < 4; ++dc) {
        union { v16h v; vi4 i[2]; } ub;
        ub.i[0] = *(const vi4*)(kp + dc * 32);
        ub.i[1] = *(const vi4*)(kp + dc * 32 + 16);
        if (nf == 0)
          c0 = __builtin_amdgcn_wmma_f32_16x16x32_f16(false, aQ[dc], false, ub.v,
                                                      (short)0, c0, false, false);
        else
          c1 = __builtin_amdgcn_wmma_f32_16x16x32_f16(false, aQ[dc], false, ub.v,
                                                      (short)0, c1, false, false);
      }
    }
#pragma unroll
    for (int r = 0; r < 8; ++r) {
      const int qrow = qBase + r + (hi ? 8 : 0);
      const int key0 = keyBase + l16;
      const int key1 = keyBase + 16 + l16;
      float s0 = c0[r] * scale + ((key0 <= qrow) ? 0.f : -65504.f);
      float s1 = c1[r] * scale + ((key1 <= qrow) ? 0.f : -65504.f);
      float v = fmaxf(s0, s1);
#pragma unroll
      for (int m = 8; m >= 1; m >>= 1) v = fmaxf(v, __shfl_xor(v, m, 32));
      const float mnew = fmaxf(mrow[r], v);
      const float corr = __expf(mrow[r] - mnew);
      mrow[r] = mnew;
      const float p0 = __expf(s0 - mnew);
      const float p1 = __expf(s1 - mnew);
      float rs = p0 + p1;
#pragma unroll
      for (int m = 8; m >= 1; m >>= 1) rs += __shfl_xor(rs, m, 32);
      lrow[r] = lrow[r] * corr + rs;
#pragma unroll
      for (int dc = 0; dc < 8; ++dc) accO[dc][r] *= corr;
      const int prow = r + (hi ? 8 : 0);
      Ps[prow * 32 + l16]      = (_Float16)p0;
      Ps[prow * 32 + 16 + l16] = (_Float16)p1;
    }
    __syncthreads();
    union { v16h v; vi4 i[2]; } up_;
    const _Float16* pp = &Ps[l16 * 32 + (hi ? 8 : 0)];
    up_.i[0] = *(const vi4*)pp;
    up_.i[1] = *(const vi4*)(pp + 16);
    const v16h aP = up_.v;
#pragma unroll
    for (int dc = 0; dc < 8; ++dc) {
      const _Float16* vp =
          VT + ((size_t)h * HD + dc * 16 + l16) * S_LEN + keyBase + (hi ? 8 : 0);
      union { v16h v; vi4 i[2]; } ub;
      ub.i[0] = *(const vi4*)vp;
      ub.i[1] = *(const vi4*)(vp + 16);
      accO[dc] = __builtin_amdgcn_wmma_f32_16x16x32_f16(false, aP, false, ub.v,
                                                        (short)0, accO[dc], false, false);
    }
    __syncthreads();
  }
#pragma unroll
  for (int dc = 0; dc < 8; ++dc)
#pragma unroll
    for (int r = 0; r < 8; ++r) {
      const int q = qBase + r + (hi ? 8 : 0);
      const float o = accO[dc][r] / lrow[r];
      O[(size_t)q * HS + h * HD + dc * 16 + l16] = (_Float16)o;
    }
}

__global__ __launch_bounds__(256) void k_quant_f16(const _Float16* __restrict__ x,
                                                   const float* __restrict__ sp,
                                                   signed char* __restrict__ o, int n4) {
  int gid = blockIdx.x * 256 + threadIdx.x;
  if (gid >= n4) return;
  const float invs = 1.f / *sp;
  int base = gid * 4, pack = 0;
#pragma unroll
  for (int j = 0; j < 4; ++j) {
    float q = rintf((float)x[base + j] * invs);
    q = fminf(fmaxf(q, -128.f), 127.f);
    pack |= ((int)q & 255) << (8 * j);
  }
  *(int*)&o[base] = pack;
}

__global__ __launch_bounds__(256) void k_silu_mul_q(const _Float16* __restrict__ g,
                                                    const float* __restrict__ u,
                                                    const float* __restrict__ sp,
                                                    signed char* __restrict__ o, int n4) {
  int gid = blockIdx.x * 256 + threadIdx.x;
  if (gid >= n4) return;
  const float invs = 1.f / *sp;
  int base = gid * 4, pack = 0;
#pragma unroll
  for (int j = 0; j < 4; ++j) {
    float hmul = (float)g[base + j] * u[base + j];
    float q = rintf(hmul * invs);
    q = fminf(fmaxf(q, -128.f), 127.f);
    pack |= ((int)q & 255) << (8 * j);
  }
  *(int*)&o[base] = pack;
}

extern "C" void kernel_launch(void* const* d_in, const int* in_sizes, int n_in,
                              void* d_out, int out_size, void* d_ws, size_t ws_size,
                              hipStream_t stream) {
  (void)in_sizes; (void)n_in; (void)out_size; (void)ws_size;
  const float* hs  = (const float*)d_in[0];
  const float* Wq = (const float*)d_in[3];  const float* bq = (const float*)d_in[4];
  const float* Wk = (const float*)d_in[5];  const float* bk = (const float*)d_in[6];
  const float* Wv = (const float*)d_in[7];  const float* bv = (const float*)d_in[8];
  const float* Wo = (const float*)d_in[9];  const float* bo = (const float*)d_in[10];
  const float* Wg = (const float*)d_in[11]; const float* bg = (const float*)d_in[12];
  const float* Wu = (const float*)d_in[13]; const float* bu = (const float*)d_in[14];
  const float* Wd = (const float*)d_in[15]; const float* bd = (const float*)d_in[16];
  const float* ln1 = (const float*)d_in[17];
  const float* ln2 = (const float*)d_in[18];
  const float* a_q = (const float*)d_in[19];
  const float* a_k = (const float*)d_in[20];
  const float* a_v = (const float*)d_in[21];
  const float* a_o = (const float*)d_in[22];
  const float* a_g = (const float*)d_in[23];
  const float* a_u = (const float*)d_in[24];
  const float* a_d = (const float*)d_in[25];
  const float* o_scale = (const float*)d_in[26];
  const float* d_scale = (const float*)d_in[27];

  char* ws = (char*)d_ws;
  size_t off = 0;
  auto alloc = [&](size_t bytes) {
    char* p = ws + off;
    off += (bytes + 255) & ~(size_t)255;
    return p;
  };
  signed char* w8q = (signed char*)alloc((size_t)HS * HS);
  signed char* w8k = (signed char*)alloc((size_t)HS * HS);
  signed char* w8v = (signed char*)alloc((size_t)HS * HS);
  signed char* w8o = (signed char*)alloc((size_t)HS * HS);
  signed char* w8g = (signed char*)alloc((size_t)IM * HS);
  signed char* w8u = (signed char*)alloc((size_t)IM * HS);
  signed char* w8d = (signed char*)alloc((size_t)HS * IM);
  signed char* x8  = (signed char*)alloc((size_t)S_LEN * HS);
  _Float16* qf   = (_Float16*)alloc((size_t)S_LEN * HS * 2);
  _Float16* kf   = (_Float16*)alloc((size_t)S_LEN * HS * 2);
  _Float16* vf   = (_Float16*)alloc((size_t)S_LEN * HS * 2);
  _Float16* vt   = (_Float16*)alloc((size_t)S_LEN * HS * 2);
  _Float16* attn = (_Float16*)alloc((size_t)S_LEN * HS * 2);
  signed char* o8 = (signed char*)alloc((size_t)S_LEN * HS);
  float* hidden = (float*)alloc((size_t)S_LEN * HS * 4);
  _Float16* gate = (_Float16*)alloc((size_t)S_LEN * IM * 2);
  float* upb = (float*)alloc((size_t)S_LEN * IM * 4);
  signed char* d8 = (signed char*)alloc((size_t)S_LEN * IM);

  const int nHH4 = HS * HS / 4;
  const int nIH4 = IM * HS / 4;
  k_quant_w<<<(nHH4 + 255) / 256, 256, 0, stream>>>(Wq, w8q, nHH4);
  k_quant_w<<<(nHH4 + 255) / 256, 256, 0, stream>>>(Wk, w8k, nHH4);
  k_quant_w<<<(nHH4 + 255) / 256, 256, 0, stream>>>(Wv, w8v, nHH4);
  k_quant_w<<<(nHH4 + 255) / 256, 256, 0, stream>>>(Wo, w8o, nHH4);
  k_quant_w<<<(nIH4 + 255) / 256, 256, 0, stream>>>(Wg, w8g, nIH4);
  k_quant_w<<<(nIH4 + 255) / 256, 256, 0, stream>>>(Wu, w8u, nIH4);
  k_quant_w<<<(nIH4 + 255) / 256, 256, 0, stream>>>(Wd, w8d, nIH4);
  k_rmsnorm_q<<<S_LEN, 256, 0, stream>>>(hs, ln1, x8);
  dim3 gHS(HS / 128, S_LEN / 128);
  k_gemm_i8<EPI_F16><<<gHS, 256, 0, stream>>>(x8, w8q, bq, a_q, nullptr, qf, S_LEN, HS, HS);
  k_gemm_i8<EPI_F16><<<gHS, 256, 0, stream>>>(x8, w8k, bk, a_k, nullptr, kf, S_LEN, HS, HS);
  k_gemm_i8<EPI_F16><<<gHS, 256, 0, stream>>>(x8, w8v, bv, a_v, nullptr, vf, S_LEN, HS, HS);
  k_rope<<<(S_LEN * NH * 64) / 256, 256, 0, stream>>>(qf, kf);
  k_vtrans<<<(S_LEN * HS) / 256, 256, 0, stream>>>(vf, vt);
  k_attn<<<dim3(S_LEN / 16, NH), 32, 0, stream>>>(qf, kf, vt, attn);
  const int nSH4 = S_LEN * HS / 4;
  k_quant_f16<<<(nSH4 + 255) / 256, 256, 0, stream>>>(attn, o_scale, o8, nSH4);
  k_gemm_i8<EPI_RES><<<gHS, 256, 0, stream>>>(o8, w8o, bo, a_o, hs, hidden, S_LEN, HS, HS);
  k_rmsnorm_q<<<S_LEN, 256, 0, stream>>>(hidden, ln2, x8);
  dim3 gIM(IM / 128, S_LEN / 128);
  k_gemm_i8<EPI_SILU><<<gIM, 256, 0, stream>>>(x8, w8g, bg, a_g, nullptr, gate, S_LEN, IM, HS);
  k_gemm_i8<EPI_F32><<<gIM, 256, 0, stream>>>(x8, w8u, bu, a_u, nullptr, upb, S_LEN, IM, HS);
  const int nSI4 = S_LEN * IM / 4;
  k_silu_mul_q<<<(nSI4 + 255) / 256, 256, 0, stream>>>(gate, upb, d_scale, d8, nSI4);
  k_gemm_i8<EPI_RES><<<gHS, 256, 0, stream>>>(d8, w8d, bd, a_d, hidden, (float*)d_out,
                                              S_LEN, HS, IM);
}
